// MyMHA_73057393705743
// MI455X (gfx1250) — compile-verified, matrix operand order checked by reading, not run on hardware
//
#include <hip/hip_runtime.h>
#include <cstdint>

// ---------------------------------------------------------------------------
// MHA forward for MI455X (gfx1250, wave32, WMMA).
// Pipeline: fp32->bf16 convert, 3x QKV GEMM (bf16 WMMA, f32 accum, async
// global->LDS double-buffered tiles), flash-attention (online softmax),
// output projection GEMM -> fp32.
// ---------------------------------------------------------------------------

typedef __attribute__((ext_vector_type(16))) __bf16    bf16x16;
typedef __attribute__((ext_vector_type(8)))  float     v8f;
typedef __attribute__((ext_vector_type(4)))  float     f32x4;
typedef __attribute__((ext_vector_type(4)))  unsigned  u32x4;
typedef __attribute__((ext_vector_type(2)))  unsigned  u32x2;

#define NB   4
#define SEQ  2048
#define NH   16
#define HD   64
#define HID  1024

// ---------------------------------------------------------------------------
// WMMA helper: D = A(16x32 bf16) x B(32x16 bf16) + C(16x16 f32)
// ---------------------------------------------------------------------------
__device__ __forceinline__ v8f wmma_bf16(bf16x16 a, bf16x16 b, v8f c) {
  return __builtin_amdgcn_wmma_f32_16x16x32_bf16(
      /*neg_a=*/false, a, /*neg_b=*/false, b,
      /*c_mod=*/(short)0, c, /*reuse_a=*/false, /*reuse_b=*/false);
}

// ---------------------------------------------------------------------------
// Fragment load for the CDNA5 16-bit A/B layout (ISA 7.12.2):
//   lane (0..15) -> row r;  lane half selects K 0-7/16-23 vs 8-15/24-31.
// Two 16-byte loads per lane (b128), works for LDS and global pointers.
// ---------------------------------------------------------------------------
__device__ __forceinline__ bf16x16 load_frag(const __bf16* base, int stride, int lane) {
  const int r  = lane & 15;
  const int hf = (lane >> 4) & 1;
  const __bf16* p = base + (size_t)r * stride + hf * 8;
  union { bf16x16 v; u32x4 u[2]; } w;
  w.u[0] = *reinterpret_cast<const u32x4*>(p);
  w.u[1] = *reinterpret_cast<const u32x4*>(p + 16);
  return w.v;
}

// ---------------------------------------------------------------------------
// Async global->LDS copy of one 32-byte (16 x bf16) row chunk per lane.
// ISA 10/15.18: GLOBAL_LOAD_ASYNC_TO_LDS_B128 vdst(lds addr), vaddr(64b), off.
// INST_OFFSET applies to both the LDS and global address, so offset:16
// fetches the second 16-byte half.  Tracked with ASYNCcnt.
// ---------------------------------------------------------------------------
__device__ __forceinline__ void async_copy32B(unsigned lds, const __bf16* g) {
  asm volatile(
      "global_load_async_to_lds_b128 %0, %1, off\n\t"
      "global_load_async_to_lds_b128 %0, %1, off offset:16"
      :: "v"(lds), "v"(g) : "memory");
}

// ---------------------------------------------------------------------------
// fp32 -> bf16 conversion (vectorized x4)
// ---------------------------------------------------------------------------
__global__ void mha_cvt_bf16(const float* __restrict__ src,
                             __bf16* __restrict__ dst, int n4) {
  int i = blockIdx.x * blockDim.x + threadIdx.x;
  if (i < n4) {
    f32x4 f = reinterpret_cast<const f32x4*>(src)[i];
    union { __bf16 e[4]; u32x2 u; } o;
    o.e[0] = (__bf16)f.x; o.e[1] = (__bf16)f.y;
    o.e[2] = (__bf16)f.z; o.e[3] = (__bf16)f.w;
    reinterpret_cast<u32x2*>(dst)[i] = o.u;
  }
}

// ---------------------------------------------------------------------------
// Tiled GEMM: C[M,N] = A[M,K](bf16) * W[N,K]^T(bf16) + bias[N](f32)
// Block tile 128x128, BK=32, 256 threads = 8 waves in a 4x2 wave grid
// (each wave: 2x4 WMMA tiles = 32x64 of C).  K-tiles are double-buffered
// in LDS and filled with async global->LDS loads (ASYNCcnt pipelined).
// mode 0: write bf16 scattered into [B,H,S,D]   (QKV projections)
// mode 1: write fp32 into [M,N] row-major        (output projection)
// ---------------------------------------------------------------------------
#define GBK 32
#define LDT 40   // padded LDS row stride (bf16 elems), keeps 16B alignment

__global__ __launch_bounds__(256) void mha_gemm_bf16(
    const __bf16* __restrict__ A,
    const __bf16* __restrict__ W,
    const float*  __restrict__ bias,
    __bf16* __restrict__ outB,
    float*  __restrict__ outF,
    int M, int N, int K, int mode)
{
  __shared__ __bf16 As[2][128][LDT];
  __shared__ __bf16 Bs[2][128][LDT];

  const int tid  = threadIdx.x;
  const int lane = tid & 31;
  const int wave = tid >> 5;
  const int wm   = wave >> 1;           // 0..3 -> M offset wm*32
  const int wn   = wave & 1;            // 0..1 -> N offset wn*64
  const int m0   = blockIdx.y * 128;
  const int n0   = blockIdx.x * 128;

  const int lr = tid >> 1;              // 0..127: tile row this thread fills
  const int lc = (tid & 1) << 4;        // 0 or 16: K-half this thread fills

  const __bf16* gA = A + (size_t)(m0 + lr) * K + lc;
  const __bf16* gB = W + (size_t)(n0 + lr) * K + lc;
  const unsigned ldsA0 = (unsigned)(uintptr_t)&As[0][lr][lc];
  const unsigned ldsA1 = (unsigned)(uintptr_t)&As[1][lr][lc];
  const unsigned ldsB0 = (unsigned)(uintptr_t)&Bs[0][lr][lc];
  const unsigned ldsB1 = (unsigned)(uintptr_t)&Bs[1][lr][lc];

  v8f acc[2][4] = {};

  // Prologue: kick off tile 0 into buffer 0 (4 async ops per wave).
  async_copy32B(ldsA0, gA);
  async_copy32B(ldsB0, gB);

  const int T = K / GBK;
  for (int t = 0; t < T; ++t) {
    const int cur = t & 1;
    if (t + 1 < T) {
      // Issue tile t+1 into the other buffer, then wait for tile t only:
      // async loads complete in order, so ASYNCcnt <= 4 means tile t landed.
      const __bf16* gA2 = gA + (size_t)(t + 1) * GBK;
      const __bf16* gB2 = gB + (size_t)(t + 1) * GBK;
      async_copy32B(cur ? ldsA0 : ldsA1, gA2);
      async_copy32B(cur ? ldsB0 : ldsB1, gB2);
      asm volatile("s_wait_asynccnt 0x4" ::: "memory");
    } else {
      asm volatile("s_wait_asynccnt 0x0" ::: "memory");
    }
    __syncthreads();   // all waves' tile-t data visible in LDS

    bf16x16 af[2], bf[4];
#pragma unroll
    for (int mt = 0; mt < 2; ++mt)
      af[mt] = load_frag(&As[cur][wm * 32 + mt * 16][0], LDT, lane);
#pragma unroll
    for (int nt = 0; nt < 4; ++nt)
      bf[nt] = load_frag(&Bs[cur][wn * 64 + nt * 16][0], LDT, lane);
#pragma unroll
    for (int mt = 0; mt < 2; ++mt)
#pragma unroll
      for (int nt = 0; nt < 4; ++nt)
        acc[mt][nt] = wmma_bf16(af[mt], bf[nt], acc[mt][nt]);

    __syncthreads();   // buffer `cur` free before tile t+2 is issued into it
  }

  // Epilogue.  C layout: VGPR i holds (M = i + 8*half, N = lane&15).
  const int r  = lane & 15;
  const int hf = lane >> 4;
#pragma unroll
  for (int mt = 0; mt < 2; ++mt) {
#pragma unroll
    for (int nt = 0; nt < 4; ++nt) {
#pragma unroll
      for (int i = 0; i < 8; ++i) {
        const int m = m0 + wm * 32 + mt * 16 + i + 8 * hf;
        const int n = n0 + wn * 64 + nt * 16 + r;
        const float v = acc[mt][nt][i] + bias[n];
        if (mode == 0) {
          const int b = m >> 11;          // m / 2048
          const int s = m & 2047;
          const int h = n >> 6;           // n / 64
          const int d = n & 63;
          outB[(((size_t)(b * NH + h) * SEQ + s) << 6) + d] = (__bf16)v;
        } else {
          outF[(size_t)m * N + n] = v;
        }
      }
    }
  }
}

// ---------------------------------------------------------------------------
// Flash attention.  grid = (SEQ/128, B*H), 256 threads = 8 waves.
// Each wave owns a 16-row query tile.  KV tiles of 128 keys.
//   S = Q(16x64) K^T  -> 8 n-tiles x 2 WMMA (K=32)
//   online softmax with 16-lane shfl_xor reductions
//   P -> per-wave LDS -> A-layout bf16 fragments
//   O += P(16x128) V(128x64) -> 4 n-tiles x 4 WMMA
// ---------------------------------------------------------------------------
__global__ __launch_bounds__(256) void mha_flash_attn(
    const __bf16* __restrict__ Q,   // [B*H, S, 64] bf16
    const __bf16* __restrict__ K,
    const __bf16* __restrict__ V,
    __bf16* __restrict__ O)         // [B, S, 1024] bf16
{
  __shared__ __bf16 Plds[8][16][136];   // per-wave P staging (34.8 KB)
  __shared__ __bf16 Vlds[64][136];      // transposed V tile   (17.4 KB)

  const int tid  = threadIdx.x;
  const int lane = tid & 31;
  const int wave = tid >> 5;
  const int r    = lane & 15;
  const int hf   = lane >> 4;
  const int bh   = blockIdx.y;          // 0..63
  const int b    = bh >> 4;
  const int h    = bh & 15;
  const int q0   = blockIdx.x * 128 + wave * 16;

  const __bf16* Qp = Q + ((size_t)bh * SEQ + q0) * HD;
  const __bf16* Kp = K + (size_t)bh * SEQ * HD;
  const __bf16* Vp = V + (size_t)bh * SEQ * HD;

  // Q fragments (K=0..31 and 32..63), loaded once from global in frag layout.
  const bf16x16 qf0 = load_frag(Qp,      HD, lane);
  const bf16x16 qf1 = load_frag(Qp + 32, HD, lane);

  v8f   oacc[4] = {};
  float mrow[8], lrow[8];
#pragma unroll
  for (int i = 0; i < 8; ++i) { mrow[i] = -1e30f; lrow[i] = 0.0f; }

  const int vj = tid >> 1;          // 0..127: V row this thread transposes
  const int vd = (tid & 1) << 5;    // 0 or 32: D-half
  const float scale = 0.125f;       // 1/sqrt(64)

  for (int kv = 0; kv < SEQ; kv += 128) {
    // --- cooperative transpose of V tile into LDS: Vlds[d][j] = V[kv+j][d]
    __syncthreads();                // previous tile's PV reads are done
    {
      const u32x4* ps = reinterpret_cast<const u32x4*>(Vp + (size_t)(kv + vj) * HD + vd);
      union { u32x4 u[4]; __bf16 e[32]; } t;
      t.u[0] = ps[0]; t.u[1] = ps[1]; t.u[2] = ps[2]; t.u[3] = ps[3];
#pragma unroll
      for (int e = 0; e < 32; ++e)
        Vlds[vd + e][vj] = t.e[e];
    }
    __syncthreads();

    // --- scores: S = Q K^T  (8 tiles of 16 keys, 2 K-steps over D)
    v8f sacc[8];
#pragma unroll
    for (int nt = 0; nt < 8; ++nt) {
      const __bf16* kp = Kp + (size_t)(kv + nt * 16) * HD;
      bf16x16 kf0 = load_frag(kp,      HD, lane);
      bf16x16 kf1 = load_frag(kp + 32, HD, lane);
      v8f s = {};
      s = wmma_bf16(qf0, kf0, s);
      s = wmma_bf16(qf1, kf1, s);
      sacc[nt] = s;
    }

    // --- online softmax (row i+8*hf lives in VGPR slot i on this lane-half)
#pragma unroll
    for (int i = 0; i < 8; ++i) {
      float mx = -1e30f;
#pragma unroll
      for (int nt = 0; nt < 8; ++nt) {
        float s = sacc[nt][i] * scale;
        sacc[nt][i] = s;
        mx = fmaxf(mx, s);
      }
#pragma unroll
      for (int msk = 1; msk < 16; msk <<= 1)
        mx = fmaxf(mx, __shfl_xor(mx, msk, 32));
      const float mnew  = fmaxf(mrow[i], mx);
      const float alpha = __expf(mrow[i] - mnew);
      mrow[i] = mnew;
      float rs = 0.0f;
#pragma unroll
      for (int nt = 0; nt < 8; ++nt) {
        float p = __expf(sacc[nt][i] - mnew);
        sacc[nt][i] = p;
        rs += p;
      }
#pragma unroll
      for (int msk = 1; msk < 16; msk <<= 1)
        rs += __shfl_xor(rs, msk, 32);
      lrow[i] = lrow[i] * alpha + rs;
#pragma unroll
      for (int nt = 0; nt < 4; ++nt)
        oacc[nt][i] *= alpha;
    }

    // --- stage P as bf16 in per-wave LDS (C layout -> row-major)
#pragma unroll
    for (int nt = 0; nt < 8; ++nt)
#pragma unroll
      for (int i = 0; i < 8; ++i)
        Plds[wave][i + 8 * hf][nt * 16 + r] = (__bf16)sacc[nt][i];

    // --- O += P V : 4 key-chunks of 32, 4 D-tiles of 16
#pragma unroll
    for (int c = 0; c < 4; ++c) {
      bf16x16 pf = load_frag(&Plds[wave][0][c * 32], 136, lane);
#pragma unroll
      for (int nt = 0; nt < 4; ++nt) {
        bf16x16 vf = load_frag(&Vlds[nt * 16][c * 32], 136, lane);
        oacc[nt] = wmma_bf16(pf, vf, oacc[nt]);
      }
    }
  }

  // --- normalize and write O in [B, S, H*D] bf16 for the final projection
#pragma unroll
  for (int nt = 0; nt < 4; ++nt)
#pragma unroll
    for (int i = 0; i < 8; ++i) {
      const int   srow = q0 + i + 8 * hf;
      const int   d    = nt * 16 + r;
      const float val  = oacc[nt][i] / lrow[i];
      O[((size_t)b * SEQ + srow) * HID + h * HD + d] = (__bf16)val;
    }
}

// ---------------------------------------------------------------------------
// Launcher
// ---------------------------------------------------------------------------
extern "C" void kernel_launch(void* const* d_in, const int* in_sizes, int n_in,
                              void* d_out, int out_size, void* d_ws, size_t ws_size,
                              hipStream_t stream) {
  (void)in_sizes; (void)n_in; (void)out_size; (void)ws_size;

  const float* x  = (const float*)d_in[0];
  const float* Wq = (const float*)d_in[1];
  const float* bq = (const float*)d_in[2];
  const float* Wk = (const float*)d_in[3];
  const float* bk = (const float*)d_in[4];
  const float* Wv = (const float*)d_in[5];
  const float* bv = (const float*)d_in[6];
  const float* Wo = (const float*)d_in[7];
  const float* bo = (const float*)d_in[8];
  float* out = (float*)d_out;

  const size_t M  = (size_t)NB * SEQ;     // 8192 tokens
  const size_t xN = M * HID;              // 8,388,608 elems
  const size_t wN = (size_t)HID * HID;    // 1,048,576 elems

  // Workspace layout (bf16): X | Wq | Wk | Wv | Wo | Q | K | V | O  (~88 MB)
  __bf16* Xb  = (__bf16*)d_ws;
  __bf16* Wqb = Xb  + xN;
  __bf16* Wkb = Wqb + wN;
  __bf16* Wvb = Wkb + wN;
  __bf16* Wvb2= Wvb;                      // alias guard (unused)
  __bf16* Wob = Wvb + wN;
  __bf16* Qb  = Wob + wN;
  __bf16* Kb  = Qb  + xN;
  __bf16* Vb  = Kb  + xN;
  __bf16* Ob  = Vb  + xN;
  (void)Wvb2;

  const int thr = 256;
  mha_cvt_bf16<<<(int)((xN / 4 + thr - 1) / thr), thr, 0, stream>>>(x,  Xb,  (int)(xN / 4));
  mha_cvt_bf16<<<(int)((wN / 4 + thr - 1) / thr), thr, 0, stream>>>(Wq, Wqb, (int)(wN / 4));
  mha_cvt_bf16<<<(int)((wN / 4 + thr - 1) / thr), thr, 0, stream>>>(Wk, Wkb, (int)(wN / 4));
  mha_cvt_bf16<<<(int)((wN / 4 + thr - 1) / thr), thr, 0, stream>>>(Wv, Wvb, (int)(wN / 4));
  mha_cvt_bf16<<<(int)((wN / 4 + thr - 1) / thr), thr, 0, stream>>>(Wo, Wob, (int)(wN / 4));

  dim3 gg(HID / 128, (unsigned)(M / 128));   // (8, 64)
  mha_gemm_bf16<<<gg, 256, 0, stream>>>(Xb, Wqb, bq, Qb, nullptr, (int)M, HID, HID, 0);
  mha_gemm_bf16<<<gg, 256, 0, stream>>>(Xb, Wkb, bk, Kb, nullptr, (int)M, HID, HID, 0);
  mha_gemm_bf16<<<gg, 256, 0, stream>>>(Xb, Wvb, bv, Vb, nullptr, (int)M, HID, HID, 0);

  dim3 ga(SEQ / 128, NB * NH);               // (16, 64)
  mha_flash_attn<<<ga, 256, 0, stream>>>(Qb, Kb, Vb, Ob);

  mha_gemm_bf16<<<gg, 256, 0, stream>>>(Ob, Wob, bo, nullptr, out, (int)M, HID, HID, 1);
}
